// CrossAttentionOflow_44779329028817
// MI455X (gfx1250) — compile-verified, matrix operand order checked by reading, not run on hardware
//
#include <hip/hip_runtime.h>
#include <hip/hip_bf16.h>
#include <math.h>
#include <stdint.h>

// ---------- problem constants ----------
#define B_    2
#define N_    2048
#define DM_   1024          // d_model = H*DH
#define H_    16
#define DH_   64
#define SCALE_  0.125f      // DH^-0.5
#define LOG2E_  1.4426950408889634f

typedef __attribute__((ext_vector_type(16))) _Float16 v16h;
typedef __attribute__((ext_vector_type(8)))  _Float16 v8h;
typedef __attribute__((ext_vector_type(8)))  float    v8f;
typedef __attribute__((ext_vector_type(4)))  unsigned int v4u;
typedef __attribute__((ext_vector_type(8)))  int      v8i;
typedef __attribute__((ext_vector_type(4)))  int      v4i;

__device__ __forceinline__ v16h cat8(v8h lo, v8h hi) {
    return __builtin_shufflevector(lo, hi, 0,1,2,3,4,5,6,7,8,9,10,11,12,13,14,15);
}

// Raw hardware transcendentals (base-2 exp, fast reciprocal).
__device__ __forceinline__ float fast_exp2(float x) { return __builtin_amdgcn_exp2f(x); }
__device__ __forceinline__ float fast_rcp(float x)  { return __builtin_amdgcn_rcpf(x); }

// DPP ROW_XMASK cross-lane move within a row of 16 lanes (VALU-only, no LDS).
template<int CTRL>
__device__ __forceinline__ float dpp_mov_f32(float x) {
    return __builtin_bit_cast(float,
        __builtin_amdgcn_update_dpp(0, __builtin_bit_cast(int, x),
                                    CTRL, 0xf, 0xf, true));
}
__device__ __forceinline__ float row16_max(float v) {
    v = fmaxf(v, dpp_mov_f32<0x161>(v));   // row_xmask:1
    v = fmaxf(v, dpp_mov_f32<0x162>(v));   // row_xmask:2
    v = fmaxf(v, dpp_mov_f32<0x164>(v));   // row_xmask:4
    v = fmaxf(v, dpp_mov_f32<0x168>(v));   // row_xmask:8
    return v;
}
__device__ __forceinline__ float row16_sum(float v) {
    v += dpp_mov_f32<0x161>(v);
    v += dpp_mov_f32<0x162>(v);
    v += dpp_mov_f32<0x164>(v);
    v += dpp_mov_f32<0x168>(v);
    return v;
}

// Load a 16x32 f16 WMMA A-fragment (or 32x16 B-fragment stored k-contiguous
// per row/column) from a padded tile.  ISA layout: lanes 0-15 hold K 0..7 and
// 16..23, lanes 16-31 hold K 8..15 and 24..31.
__device__ __forceinline__ v16h load_frag_lds(const _Float16* p, int row_stride,
                                              int row, int kbase, int lhalf) {
    const _Float16* q = p + (size_t)row * row_stride + kbase + lhalf * 8;
    v8h lo = *(const v8h*)(q);
    v8h hi = *(const v8h*)(q + 16);
    return cat8(lo, hi);
}
__device__ __forceinline__ v16h load_frag_global(const _Float16* p, int row_stride,
                                                 int row, int kbase, int lhalf) {
    const _Float16* q = p + (size_t)row * row_stride + kbase + lhalf * 8;
    v8h lo = *(const v8h*)(q);
    v8h hi = *(const v8h*)(q + 16);
    return cat8(lo, hi);
}

// ======================================================================
// Tiled GEMM:  C[M x Nout] = A[M x K] @ W[K x Nout] * scale (+ bias)
// Block: 256 threads = 8 waves; tile 128x128, K-step 32.
// Wave w computes a 32x64 sub-tile: rows (w&3)*32, cols (w>>2)*64.
// ======================================================================
#define BM 128
#define BN 128
#define BK 32
#define LDA 40   // f16 elements; 80B row stride (16B aligned)
#define LDB 40

template<bool A_F16, bool OUT_F16, bool BIAS>
__global__ __launch_bounds__(256)
void gemm_wmma_kernel(const void* __restrict__ Ap,
                      const float* __restrict__ W,
                      const float* __restrict__ bias,
                      void* __restrict__ Cp,
                      int M, int K, int Nout, float scale)
{
    __shared__ _Float16 Asm[BM * LDA];
    __shared__ _Float16 Bsm[BN * LDB];   // transposed: [n][k], k contiguous

    const int tid   = threadIdx.x;
    const int wave  = tid >> 5;
    const int lane  = tid & 31;
    const int l16   = lane & 15;
    const int lhalf = lane >> 4;

    const int tileM = blockIdx.y * BM;
    const int tileN = blockIdx.x * BN;
    const int wr = wave & 3;    // row group  (32 rows)
    const int wc = wave >> 2;   // col group  (64 cols)

    v8f acc[2][4];
#pragma unroll
    for (int i = 0; i < 2; ++i)
#pragma unroll
        for (int j = 0; j < 4; ++j)
            acc[i][j] = v8f{0.f,0.f,0.f,0.f,0.f,0.f,0.f,0.f};

    for (int k0 = 0; k0 < K; k0 += BK) {
        // ---- stage A tile (128 x 32) as f16 ----
        {
            const int r = tid >> 1;
            const int c = (tid & 1) * 16;
            const int arow = tileM + r;
            if (A_F16) {
                const _Float16* Af = (const _Float16*)Ap + (size_t)arow * K + k0 + c;
                *(v8h*)(Asm + r * LDA + c)     = *(const v8h*)(Af);
                *(v8h*)(Asm + r * LDA + c + 8) = *(const v8h*)(Af + 8);
            } else {
                const float* Af = (const float*)Ap + (size_t)arow * K + k0 + c;
#pragma unroll
                for (int u = 0; u < 16; ++u)
                    Asm[r * LDA + c + u] = (_Float16)Af[u];
                if (k0 + BK < K)
                    __builtin_prefetch(Af + BK, 0, 0);   // global_prefetch_b8
            }
        }
        // ---- stage W tile (32 x 128) transposed into Bsm[n][k] ----
        {
            const int kk = tid >> 3;           // 0..31
            const int nb = (tid & 7) * 16;     // 0..112
            const float* src = W + (size_t)(k0 + kk) * Nout + tileN + nb;
#pragma unroll
            for (int u = 0; u < 16; ++u)
                Bsm[(nb + u) * LDB + kk] = (_Float16)src[u];
            if (k0 + BK < K)
                __builtin_prefetch(src + (size_t)BK * Nout, 0, 0);
        }
        __syncthreads();

        // ---- 8 WMMAs per wave ----
#pragma unroll
        for (int i = 0; i < 2; ++i) {
            v16h af = load_frag_lds(Asm, LDA, wr * 32 + i * 16 + l16, 0, lhalf);
#pragma unroll
            for (int j = 0; j < 4; ++j) {
                v16h bf = load_frag_lds(Bsm, LDB, wc * 64 + j * 16 + l16, 0, lhalf);
                acc[i][j] = __builtin_amdgcn_wmma_f32_16x16x32_f16(
                    false, af, false, bf, (short)0, acc[i][j], false, false);
            }
        }
        __syncthreads();
    }

    // ---- epilogue: C layout lanes 0-15 -> M=r, lanes 16-31 -> M=r+8 ----
#pragma unroll
    for (int i = 0; i < 2; ++i) {
#pragma unroll
        for (int j = 0; j < 4; ++j) {
            const int col = tileN + wc * 64 + j * 16 + l16;
            const int rb  = tileM + wr * 32 + i * 16 + lhalf * 8;
#pragma unroll
            for (int r = 0; r < 8; ++r) {
                float v = acc[i][j][r] * scale;
                if (BIAS) v += bias[col];
                const size_t idx = (size_t)(rb + r) * Nout + col;
                if (OUT_F16) ((_Float16*)Cp)[idx] = (_Float16)v;
                else         ((float*)Cp)[idx]    = v;
            }
        }
    }
}

// ======================================================================
// Flash attention, one (b,h) per blockIdx.y, 128 query rows per block.
// Q/K/V are f16 in ws with row stride DM_ and head offset h*DH_.
// K already folded with SCALE*log2(e) -> softmax uses raw v_exp_f32.
// K tile (64x64, padded rows) is staged by the Tensor Data Mover.
// ======================================================================
#define KB    64   // keys per block-iteration
#define KT_LD 72   // Ktile [64][64] stride: 128B data + 16B pad per row
#define VT_LD 72   // Vt    [64][64] stride (keys contiguous)
#define PL_LD 72   // P     [16][64] stride per wave

// TDM 2D descriptor: load a 64x64 f16 tile (row stride DM_ elements) from
// global into LDS at lds_off, padding each 128B row with 16B (-> KT_LD rows).
__device__ __forceinline__ void tdm_load_k_tile(const _Float16* gsrc,
                                                unsigned int lds_off) {
    const unsigned long long ga = (unsigned long long)(uintptr_t)gsrc;
    v4u g0;
    g0.x = 1u;                                    // count=1 (valid user D#)
    g0.y = lds_off;                               // lds_addr
    g0.z = (unsigned int)ga;                      // global_addr[31:0]
    g0.w = (unsigned int)(ga >> 32) | 0x80000000u;// global_addr[56:32] | type=2
    v8i g1;
    g1[0] = (int)((1u << 16)                      // data_size = 2 bytes
                | (1u << 20)                      // pad_enable
                | (4u << 22)                      // pad_interval: 32 dwords
                | (3u << 25));                    // pad_amount:   4 dwords
    g1[1] = (int)(64u << 16);                     // tensor_dim0 = 64
    g1[2] = (int)(64u << 16);                     // tensor_dim1 = 64
    g1[3] = (int)(64u << 16);                     // tile_dim0 = 64
    g1[4] = 64;                                   // tile_dim1 = 64, tile_dim2 = 0
    g1[5] = DM_;                                  // tensor_dim0_stride = 1024
    g1[6] = 0;
    g1[7] = 0;
    const v4i z4 = {0, 0, 0, 0};                  // groups 2/3 unused (2D)
    const v8i z8 = {0, 0, 0, 0, 0, 0, 0, 0};
    __builtin_amdgcn_tensor_load_to_lds(g0, g1, z4, z4, z8, 0);
}

__global__ __launch_bounds__(256)
void flash_attn_kernel(const _Float16* __restrict__ Q,
                       const _Float16* __restrict__ Kp,
                       const _Float16* __restrict__ V,
                       _Float16* __restrict__ O)
{
    __shared__ __attribute__((aligned(16))) _Float16 Kt[KB * KT_LD];   // [key][d]
    __shared__ __attribute__((aligned(16))) _Float16 Vt[DH_ * VT_LD];  // [d][key]
    __shared__ __attribute__((aligned(16))) _Float16 Pl[8][16 * PL_LD];

    const int tid   = threadIdx.x;
    const int wave  = tid >> 5;
    const int lane  = tid & 31;
    const int l16   = lane & 15;
    const int lhalf = lane >> 4;

    const int bh = blockIdx.y;
    const int b  = bh / H_;
    const int h  = bh % H_;
    const size_t base = (size_t)b * N_ * DM_ + (size_t)h * DH_;

    const unsigned int kt_lds = (unsigned int)(uintptr_t)(&Kt[0]); // LDS offset
    const int qrow = blockIdx.x * 128 + wave * 16;   // first query row of wave

    // Q fragments (16 x 64 = two 16x32 A-frags), loaded once.
    v16h qf[2];
#pragma unroll
    for (int dk = 0; dk < 2; ++dk)
        qf[dk] = load_frag_global(Q + base, DM_, qrow + l16, dk * 32, lhalf);

    float mrow[8], lrow[8];
#pragma unroll
    for (int r = 0; r < 8; ++r) { mrow[r] = -1e30f; lrow[r] = 0.f; }
    v8f acc[4];
#pragma unroll
    for (int j = 0; j < 4; ++j) acc[j] = v8f{0.f,0.f,0.f,0.f,0.f,0.f,0.f,0.f};

    for (int kb = 0; kb < N_; kb += KB) {
        // ---- K tile via Tensor Data Mover (wave 0 issues + waits) ----
        if (wave == 0) {
            tdm_load_k_tile(Kp + base + (size_t)kb * DM_, kt_lds);
            __builtin_amdgcn_s_wait_tensorcnt(0);
        }
        // ---- cooperative stage of V^T (64 keys x 64 d, transposed) ----
        {
            const int r = tid >> 2;            // key row 0..63
            const int c = (tid & 3) * 16;      // d base 0,16,32,48
            const _Float16* vp = V + base + (size_t)(kb + r) * DM_ + c;
            v8h v0 = *(const v8h*)(vp);
            v8h v1 = *(const v8h*)(vp + 8);
#pragma unroll
            for (int u = 0; u < 8; ++u) {
                Vt[(c + u) * VT_LD + r]     = v0[u];
                Vt[(c + 8 + u) * VT_LD + r] = v1[u];
            }
            if (kb + KB < N_)
                __builtin_prefetch(vp + (size_t)KB * DM_, 0, 0);
        }
        __syncthreads();

        // ---- S = Q @ K^T  (16 x 64, four 16x16 C-frags) ----
        v8f s[4];
#pragma unroll
        for (int j = 0; j < 4; ++j) s[j] = v8f{0.f,0.f,0.f,0.f,0.f,0.f,0.f,0.f};
#pragma unroll
        for (int j = 0; j < 4; ++j)
#pragma unroll
            for (int dk = 0; dk < 2; ++dk) {
                v16h bf = load_frag_lds(Kt, KT_LD, j * 16 + l16, dk * 32, lhalf);
                s[j] = __builtin_amdgcn_wmma_f32_16x16x32_f16(
                    false, qf[dk], false, bf, (short)0, s[j], false, false);
            }

        // ---- online softmax (DPP row reductions, raw v_exp_f32) ----
#pragma unroll
        for (int r = 0; r < 8; ++r) {
            float mx = fmaxf(fmaxf(s[0][r], s[1][r]), fmaxf(s[2][r], s[3][r]));
            mx = row16_max(mx);
            const float mnew  = fmaxf(mrow[r], mx);
            const float alpha = fast_exp2(mrow[r] - mnew);
            float rs = 0.f;
#pragma unroll
            for (int j = 0; j < 4; ++j) {
                const float p = fast_exp2(s[j][r] - mnew);
                s[j][r] = p;
                rs += p;
            }
            rs = row16_sum(rs);
            lrow[r] = lrow[r] * alpha + rs;
            mrow[r] = mnew;
#pragma unroll
            for (int j = 0; j < 4; ++j)
                acc[j][r] *= alpha;
        }

        // ---- P: C-layout -> A-layout via per-wave LDS bounce ----
        _Float16* pw = &Pl[wave][0];
#pragma unroll
        for (int r = 0; r < 8; ++r) {
            const int row = lhalf * 8 + r;
#pragma unroll
            for (int j = 0; j < 4; ++j)
                pw[row * PL_LD + j * 16 + l16] = (_Float16)s[j][r];
        }
        v16h pf[2];
#pragma unroll
        for (int kk = 0; kk < 2; ++kk)
            pf[kk] = load_frag_lds(pw, PL_LD, l16, kk * 32, lhalf);

        // ---- O += P @ V  (four 16x16 col-frags over DH=64, k = 64 keys) ----
#pragma unroll
        for (int j = 0; j < 4; ++j)
#pragma unroll
            for (int kk = 0; kk < 2; ++kk) {
                v16h vf = load_frag_lds(Vt, VT_LD, j * 16 + l16, kk * 32, lhalf);
                acc[j] = __builtin_amdgcn_wmma_f32_16x16x32_f16(
                    false, pf[kk], false, vf, (short)0, acc[j], false, false);
            }
        __syncthreads();
    }

    // ---- normalize and store O as f16 [B,N,H*DH] ----
#pragma unroll
    for (int r = 0; r < 8; ++r) {
        const float inv = fast_rcp(lrow[r]);
        const int row = qrow + lhalf * 8 + r;
#pragma unroll
        for (int j = 0; j < 4; ++j) {
            const int col = j * 16 + l16;
            O[base + (size_t)row * DM_ + col] = (_Float16)(acc[j][r] * inv);
        }
    }
}

// ======================================================================
// launcher
// ======================================================================
extern "C" void kernel_launch(void* const* d_in, const int* in_sizes, int n_in,
                              void* d_out, int out_size, void* d_ws, size_t ws_size,
                              hipStream_t stream) {
    const float* x   = (const float*)d_in[0];
    const float* ctx = (const float*)d_in[1];
    const float* Wq  = (const float*)d_in[2];
    const float* Wk  = (const float*)d_in[3];
    const float* Wv  = (const float*)d_in[4];
    const float* Wo  = (const float*)d_in[5];
    const float* bo  = (const float*)d_in[6];

    const size_t elems = (size_t)B_ * N_ * DM_;   // 4096*1024
    _Float16* qws = (_Float16*)d_ws;
    _Float16* kws = qws + elems;
    _Float16* vws = kws + elems;
    _Float16* ows = vws + elems;

    const int M = B_ * N_;        // 4096
    dim3 blk(256);
    dim3 gG(DM_ / BN, M / BM);    // (8, 32)

    // Q = x @ Wq ; K = (ctx @ Wk) * SCALE*log2(e) ; V = ctx @ Wv  -> f16 ws
    gemm_wmma_kernel<false, true, false><<<gG, blk, 0, stream>>>(
        x,   Wq, nullptr, qws, M, DM_, DM_, 1.0f);
    gemm_wmma_kernel<false, true, false><<<gG, blk, 0, stream>>>(
        ctx, Wk, nullptr, kws, M, DM_, DM_, SCALE_ * LOG2E_);
    gemm_wmma_kernel<false, true, false><<<gG, blk, 0, stream>>>(
        ctx, Wv, nullptr, vws, M, DM_, DM_, 1.0f);

    // attention per (b,h)
    dim3 gA(N_ / 128, B_ * H_);   // (16, 32)
    flash_attn_kernel<<<gA, blk, 0, stream>>>(qws, kws, vws, ows);

    // out = O @ Wo + bo  -> f32 d_out
    gemm_wmma_kernel<true, false, true><<<gG, blk, 0, stream>>>(
        ows, Wo, bo, d_out, M, DM_, DM_, 1.0f);
}
